// MultiHeadedAttention_42640435315397
// MI455X (gfx1250) — compile-verified, matrix operand order checked by reading, not run on hardware
//
#include <hip/hip_runtime.h>

// ---------------------------------------------------------------------------
// MI455X (gfx1250) windowed multi-head attention block.
// All GEMM stages run on v_wmma_f32_16x16x32_bf16 (wave32, f32 accumulate).
// Every fragment is fed by contiguous 16-byte loads; weights/activations are
// pre-packed to bf16 in K-major layouts so hot loops have no conversion VALU.
// GEMM waves use 2- or 4-way register blocking: one A fragment feeds 2 (QKV) or 4 WMMAs.
// ---------------------------------------------------------------------------

typedef __attribute__((ext_vector_type(16))) __bf16 v16bf;
typedef __attribute__((ext_vector_type(8)))  __bf16 v8bf;
typedef __attribute__((ext_vector_type(8)))  float  v8f;
typedef __attribute__((ext_vector_type(4)))  float  v4f;

#define BT      16      // batch*time
#define BATCH   2
#define TFR     8       // time frames per batch
#define CI      128     // input channels
#define DM      256     // model channels
#define HW      64
#define PIX     4096    // 64*64
// chunk0: 8x8 windows -> N=512 tokens/batch, D=8192
// chunk1: 4x4 windows -> N=2048 tokens/batch, D=2048
// feature index inside a chunk: f = ((h%win)*win + (w%win))*128 + dk
// (fixed permutation of the contraction axis, identical for Q/K/V, so the
//  attention math is unchanged but PV output becomes channel-contiguous)

__device__ __forceinline__ float bf2f(__bf16 v) {
  unsigned short h = __builtin_bit_cast(unsigned short, v);
  unsigned u = ((unsigned)h) << 16;
  return __builtin_bit_cast(float, u);
}
__device__ __forceinline__ __bf16 f2bf(float f) {
  unsigned u = __builtin_bit_cast(unsigned, f);
  unsigned r = (u + 0x7FFFu + ((u >> 16) & 1u)) >> 16;   // round-nearest-even
  unsigned short h = (unsigned short)r;
  return __builtin_bit_cast(__bf16, h);
}
__device__ __forceinline__ v8bf cvt8(v8f a) {
  v8bf r;
#pragma unroll
  for (int i = 0; i < 8; ++i) r[i] = f2bf(a[i]);
  return r;
}

// A/B fragment load for 16-bit WMMA operands (ISA 7.12.2): lane-half hi needs
// K = hi*8..hi*8+7 and 16+hi*8..16+hi*8+7  ->  two contiguous b128 loads.
__device__ __forceinline__ v16bf frag_ld(const __bf16* __restrict__ row, int hi) {
  v8bf lo = *(const v8bf*)(row + hi * 8);
  v8bf hh = *(const v8bf*)(row + 16 + hi * 8);
  return __builtin_shufflevector(lo, hh, 0, 1, 2, 3, 4, 5, 6, 7,
                                 8, 9, 10, 11, 12, 13, 14, 15);
}

#define WMMA_BF16(A, B, C) \
  __builtin_amdgcn_wmma_f32_16x16x32_bf16(false, (A), false, (B), (short)0, (C), false, false)

// ---------------------------------------------------------------------------
// Prep 1: pack weights to bf16.  wqkv: [3][256][128].  wot: [9][co=256][ci=256]
// (conv weights transposed tap-major so B-fragment rows are ci-contiguous).
// ---------------------------------------------------------------------------
__global__ void prep_weights_kernel(const float* __restrict__ wq,
                                    const float* __restrict__ wk,
                                    const float* __restrict__ wv,
                                    const float* __restrict__ wo,
                                    __bf16* __restrict__ wqkv,
                                    __bf16* __restrict__ wot) {
  const int i = blockIdx.x * blockDim.x + threadIdx.x;
  if (i < 3 * DM * CI) {
    const int op = i / (DM * CI), r = i % (DM * CI);
    const float* w = (op == 0) ? wq : (op == 1) ? wk : wv;
    wqkv[i] = f2bf(w[r]);
  }
  if (i < 9 * DM * DM) {
    const int tap = i / (DM * DM), rem = i % (DM * DM);
    const int co = rem / DM, ci = rem % DM;
    wot[i] = f2bf(wo[(co * DM + ci) * 9 + tap]);
  }
}

// ---------------------------------------------------------------------------
// Prep 2: transpose x [bt][128][64][64] f32 -> xt [pixel=65536][128] bf16
// via 32x32 LDS tiles (coalesced both directions).
// ---------------------------------------------------------------------------
__global__ void xt_kernel(const float* __restrict__ x, __bf16* __restrict__ xt) {
  __shared__ float tile[32][33];
  const int ct = blockIdx.x & 3;        // 128/32 channel tiles
  const int pt = blockIdx.x >> 2;       // 65536/32 pixel tiles
  const int c0 = ct << 5, p0 = pt << 5;
  const int bti = p0 >> 12, pin = p0 & 4095;
#pragma unroll
  for (int rr = 0; rr < 4; ++rr) {
    const int cl = (threadIdx.y << 2) + rr;
    tile[cl][threadIdx.x] =
        x[((size_t)(bti * CI + c0 + cl) << 12) + pin + threadIdx.x];
  }
  __syncthreads();
#pragma unroll
  for (int rr = 0; rr < 4; ++rr) {
    const int pl = (threadIdx.y << 2) + rr;
    xt[(size_t)(p0 + pl) * CI + c0 + threadIdx.x] = f2bf(tile[threadIdx.x][pl]);
  }
}

// ---------------------------------------------------------------------------
// Stage 1: QKV projection. GEMM Wbf[3*256,128] x Xt[65536,128]^T.
// One wave: 16(out-ch) x 32(pixels, 2 tiles sharing the A fragment).
//   Q: [b][N][D]   (pre-scaled 1/sqrt(D))     K: [b][N][D]     V^T: [b][D][N]
// ---------------------------------------------------------------------------
__global__ void qkv_proj_kernel(const __bf16* __restrict__ xt,
                                const __bf16* __restrict__ wqkv,
                                const float* __restrict__ bq,
                                const float* __restrict__ bk,
                                const float* __restrict__ bv,
                                __bf16* __restrict__ qbuf,
                                __bf16* __restrict__ kbuf,
                                __bf16* __restrict__ vtbuf) {
  const int lane = threadIdx.x & 31;
  const int wave = blockIdx.x * (blockDim.x >> 5) + (threadIdx.x >> 5);
  const int total = 3 * 16 * 2048;
  if (wave >= total) return;
  const int pp = wave & 2047;           // pixel pair-tile (32 pixels)
  const int tmp = wave >> 11;
  const int mo = tmp & 15;              // out-channel tile
  const int op = tmp >> 4;              // 0=q 1=k 2=v
  const float* bias = (op == 0) ? bq : (op == 1) ? bk : bv;

  const int hi = lane >> 4, lm = lane & 15;
  const __bf16* wrow = wqkv + ((size_t)op * DM + mo * 16 + lm) * CI;
  const __bf16* xrow0 = xt + (size_t)((pp << 5) + lm) * CI;
  const __bf16* xrow1 = xrow0 + (size_t)16 * CI;

  v8f acc0 = {}, acc1 = {};
#pragma unroll
  for (int kc = 0; kc < CI; kc += 32) {
    const v16bf a  = frag_ld(wrow + kc, hi);
    const v16bf b0 = frag_ld(xrow0 + kc, hi);
    const v16bf b1 = frag_ld(xrow1 + kc, hi);
    acc0 = WMMA_BF16(a, b0, acc0);
    acc1 = WMMA_BF16(a, b1, acc1);
  }

#pragma unroll
  for (int r = 0; r < 8; ++r) {
    const int d  = mo * 16 + r + hi * 8;         // out channel 0..255
    const int c  = d >> 7;                       // chunk id
    const int dk = d & 127;
    const int wl = c ? 2 : 3;                    // log2(window)
    const int ol = c ? 4 : 3;                    // log2(out_h)
    const int Nc = c ? 2048 : 512;
    const int Dc = c ? 2048 : 8192;
    const size_t cbase = c ? (size_t)BATCH * 512 * 8192 : 0;
    const int winm = (1 << wl) - 1;
    const float bval = bias[d];
    const float scale = c ? 0.02209708691f : 0.01104854346f;   // 1/sqrt(D)
#pragma unroll
    for (int t = 0; t < 2; ++t) {
      const int p = (pp << 5) + t * 16 + lm;     // pixel column
      const int bti = p >> 12, h = (p >> 6) & 63, w = p & 63;
      const int bi = bti >> 3, ti = bti & 7;
      const int n = (ti << (2 * ol)) + ((h >> wl) << ol) + (w >> wl);
      const int pos = ((h & winm) << wl) + (w & winm);
      const int f = (pos << 7) + dk;             // pos-major feature index
      const float val = (t ? acc1[r] : acc0[r]) + bval;
      if (op == 0)      qbuf[cbase + (size_t)(bi * Nc + n) * Dc + f] = f2bf(val * scale);
      else if (op == 1) kbuf[cbase + (size_t)(bi * Nc + n) * Dc + f] = f2bf(val);
      else              vtbuf[cbase + (size_t)(bi * Dc + f) * Nc + n] = f2bf(val);
    }
  }
}

// ---------------------------------------------------------------------------
// Stage 2: S = Q.K^T computed as S^T tiles: A = K rows (m), B = Q rows (n),
// 4 query tiles share the A fragment (5 fragment loads -> 4 WMMAs/K-step).
// Store: per lane 8 consecutive m -> one v8bf store per tile into S[n][m].
// ---------------------------------------------------------------------------
__global__ void attn_scores_kernel(const __bf16* __restrict__ q,
                                   const __bf16* __restrict__ k,
                                   __bf16* __restrict__ s,
                                   int N, int D) {
  const int lane = threadIdx.x & 31;
  const int wave = blockIdx.x * (blockDim.x >> 5) + (threadIdx.x >> 5);
  const int nt = N >> 4, ntq = nt >> 2;
  const int total = BATCH * nt * ntq;
  if (wave >= total) return;
  const int bi  = wave / (nt * ntq);
  const int rem = wave % (nt * ntq);
  const int m0 = (rem / ntq) << 4;       // key tile (rows)
  const int n0 = (rem % ntq) << 6;       // query quad-tile (cols)
  const int hi = lane >> 4, lm = lane & 15;

  const __bf16* krow  = k + (size_t)(bi * N + m0 + lm) * D;
  const __bf16* qrow0 = q + (size_t)(bi * N + n0 + lm) * D;
  const size_t qstride = (size_t)16 * D;

  v8f acc0 = {}, acc1 = {}, acc2 = {}, acc3 = {};
  for (int kc = 0; kc < D; kc += 32) {
    __builtin_prefetch(krow + kc + 256, 0, 1);
    const v16bf a  = frag_ld(krow + kc, hi);
    const v16bf b0 = frag_ld(qrow0 + kc, hi);
    const v16bf b1 = frag_ld(qrow0 + qstride + kc, hi);
    const v16bf b2 = frag_ld(qrow0 + 2 * qstride + kc, hi);
    const v16bf b3 = frag_ld(qrow0 + 3 * qstride + kc, hi);
    acc0 = WMMA_BF16(a, b0, acc0);
    acc1 = WMMA_BF16(a, b1, acc1);
    acc2 = WMMA_BF16(a, b2, acc2);
    acc3 = WMMA_BF16(a, b3, acc3);
  }
  const size_t soff = (size_t)(bi * N + n0 + lm) * N + m0 + hi * 8;
  *(v8bf*)(s + soff)                   = cvt8(acc0);
  *(v8bf*)(s + soff + (size_t)16 * N)  = cvt8(acc1);
  *(v8bf*)(s + soff + (size_t)32 * N)  = cvt8(acc2);
  *(v8bf*)(s + soff + (size_t)48 * N)  = cvt8(acc3);
}

// ---------------------------------------------------------------------------
// Stage 3: row softmax, in place on the bf16 score buffer. One block per row.
// ---------------------------------------------------------------------------
__global__ void softmax_kernel(__bf16* __restrict__ p, int N) {
  __bf16* pr = p + (size_t)blockIdx.x * N;
  const int tid = threadIdx.x;
  const int nw  = blockDim.x >> 5;
  __shared__ float red[8];

  float lmax = -3.0e38f;
  for (int i = tid; i < N; i += blockDim.x) lmax = fmaxf(lmax, bf2f(pr[i]));
  for (int o = 16; o > 0; o >>= 1) lmax = fmaxf(lmax, __shfl_xor(lmax, o, 32));
  if ((tid & 31) == 0) red[tid >> 5] = lmax;
  __syncthreads();
  float m = red[0];
  for (int i = 1; i < nw; ++i) m = fmaxf(m, red[i]);
  __syncthreads();

  float lsum = 0.f;
  for (int i = tid; i < N; i += blockDim.x) lsum += __expf(bf2f(pr[i]) - m);
  for (int o = 16; o > 0; o >>= 1) lsum += __shfl_xor(lsum, o, 32);
  if ((tid & 31) == 0) red[tid >> 5] = lsum;
  __syncthreads();
  float ssum = 0.f;
  for (int i = 0; i < nw; ++i) ssum += red[i];
  const float inv = 1.0f / ssum;

  for (int i = tid; i < N; i += blockDim.x)
    pr[i] = f2bf(__expf(bf2f(pr[i]) - m) * inv);
}

// ---------------------------------------------------------------------------
// Stage 4: Y = P.V computed as Y^T tiles: A = V^T rows (f), B = P rows (n),
// 4 token tiles share the A fragment.  Per lane: 8 consecutive features =
// 8 consecutive NHWC channels -> one v8bf store per tile.
// ---------------------------------------------------------------------------
__global__ void attn_pv_kernel(const __bf16* __restrict__ pmat,
                               const __bf16* __restrict__ vt,
                               __bf16* __restrict__ ybf,   // NHWC [bt][64][64][256]
                               int N, int D, int chunk) {
  const int lane = threadIdx.x & 31;
  const int wave = blockIdx.x * (blockDim.x >> 5) + (threadIdx.x >> 5);
  const int ft = D >> 4, ntq = N >> 6;
  const int total = BATCH * ft * ntq;
  if (wave >= total) return;
  const int bi  = wave / (ft * ntq);
  const int rem = wave % (ft * ntq);
  const int f0 = (rem / ntq) << 4;       // feature tile (rows)
  const int n0 = (rem % ntq) << 6;       // token quad-tile (cols)
  const int hi = lane >> 4, lm = lane & 15;

  const __bf16* vrow  = vt + (size_t)(bi * D + f0 + lm) * N;
  const __bf16* prow0 = pmat + (size_t)(bi * N + n0 + lm) * N;
  const size_t pstride = (size_t)16 * N;

  v8f acc0 = {}, acc1 = {}, acc2 = {}, acc3 = {};
  for (int kc = 0; kc < N; kc += 32) {
    __builtin_prefetch(vrow + kc + 256, 0, 1);
    const v16bf a  = frag_ld(vrow + kc, hi);
    const v16bf b0 = frag_ld(prow0 + kc, hi);
    const v16bf b1 = frag_ld(prow0 + pstride + kc, hi);
    const v16bf b2 = frag_ld(prow0 + 2 * pstride + kc, hi);
    const v16bf b3 = frag_ld(prow0 + 3 * pstride + kc, hi);
    acc0 = WMMA_BF16(a, b0, acc0);
    acc1 = WMMA_BF16(a, b1, acc1);
    acc2 = WMMA_BF16(a, b2, acc2);
    acc3 = WMMA_BF16(a, b3, acc3);
  }

  const int wl = chunk ? 2 : 3, ol = chunk ? 4 : 3;
  const int f  = f0 + hi * 8;                    // first of 8 consecutive feats
  const int pos = f >> 7, dk0 = f & 127;
  const int ph = pos >> wl, pw = pos & ((1 << wl) - 1);
  const int d0 = (chunk << 7) + dk0;             // NHWC channel base
#pragma unroll
  for (int t = 0; t < 4; ++t) {
    const int n  = n0 + t * 16 + lm;
    const int ti = n >> (2 * ol);
    const int remn = n & ((1 << (2 * ol)) - 1);
    const int bh = remn >> ol, bw = remn & ((1 << ol) - 1);
    const int hh = (bh << wl) + ph, ww = (bw << wl) + pw;
    const int bti = bi * TFR + ti;
    const size_t addr = ((size_t)(((bti << 6) + hh) << 6) + ww) * DM + d0;
    const v8f acc = (t == 0) ? acc0 : (t == 1) ? acc1 : (t == 2) ? acc2 : acc3;
    *(v8bf*)(ybf + addr) = cvt8(acc);
  }
}

// ---------------------------------------------------------------------------
// Stage 5: 3x3 SAME conv as implicit GEMM on NHWC input:
// A = input patches (pixel rows, ci-contiguous), B = wot[tap] (co rows),
// 4 co tiles share the bounds-selected A fragment (9 taps x 8 K x 4 = 288
// WMMAs/wave).  Output NCHW f32 with bias + LeakyReLU; per lane 8 consecutive
// pixels -> two float4 stores per tile.
// ---------------------------------------------------------------------------
__global__ void conv3x3_kernel(const __bf16* __restrict__ y,      // NHWC bf16
                               const __bf16* __restrict__ wot,    // [9][256][256]
                               const float* __restrict__ bo,
                               float* __restrict__ out) {
  const int lane = threadIdx.x & 31;
  const int wave = blockIdx.x * (blockDim.x >> 5) + (threadIdx.x >> 5);
  if (wave >= 4096 * 4) return;
  const int pt = wave >> 2;            // pixel tile (16 pixels, M rows)
  const int co0 = (wave & 3) << 6;     // co quad-tile (N cols)
  const int hi = lane >> 4, lm = lane & 15;
  const int p = (pt << 4) + lm;        // this lane's A-row pixel
  const int bti = p >> 12, h = (p >> 6) & 63, w = p & 63;

  v16bf zfrag;
#pragma unroll
  for (int i = 0; i < 16; ++i) zfrag[i] = f2bf(0.f);

  v8f acc0 = {}, acc1 = {}, acc2 = {}, acc3 = {};
  for (int kh = 0; kh < 3; ++kh) {
    const int hh = h + kh - 1;
    const bool hin = (hh >= 0) && (hh < HW);
    for (int kw = 0; kw < 3; ++kw) {
      const int ww = w + kw - 1;
      const bool in = hin && (ww >= 0) && (ww < HW);
      const size_t pix = in ? ((size_t)(((bti << 6) + hh) << 6) + ww) : 0;
      const __bf16* arow = y + pix * DM;
      const int tap = kh * 3 + kw;
      const __bf16* wt0 = wot + ((size_t)tap * DM + co0 + lm) * DM;
      const size_t wstride = (size_t)16 * DM;
#pragma unroll
      for (int cc = 0; cc < DM; cc += 32) {
        const v16bf al = frag_ld(arow + cc, hi);
        const v16bf a  = in ? al : zfrag;
        const v16bf b0 = frag_ld(wt0 + cc, hi);
        const v16bf b1 = frag_ld(wt0 + wstride + cc, hi);
        const v16bf b2 = frag_ld(wt0 + 2 * wstride + cc, hi);
        const v16bf b3 = frag_ld(wt0 + 3 * wstride + cc, hi);
        acc0 = WMMA_BF16(a, b0, acc0);
        acc1 = WMMA_BF16(a, b1, acc1);
        acc2 = WMMA_BF16(a, b2, acc2);
        acc3 = WMMA_BF16(a, b3, acc3);
      }
    }
  }
  // store: lane col = co, rows = 8 consecutive pixels (p0 + hi*8 + 0..7)
  const int pin0 = ((pt << 4) & 4095) + hi * 8;
#pragma unroll
  for (int t = 0; t < 4; ++t) {
    const int co = co0 + t * 16 + lm;
    const float bval = bo[co];
    const v8f acc = (t == 0) ? acc0 : (t == 1) ? acc1 : (t == 2) ? acc2 : acc3;
    v4f lo, hi4;
#pragma unroll
    for (int i = 0; i < 4; ++i) {
      float v0 = acc[i] + bval;     lo[i]  = (v0 >= 0.f) ? v0 : 0.2f * v0;
      float v1 = acc[i + 4] + bval; hi4[i] = (v1 >= 0.f) ? v1 : 0.2f * v1;
    }
    float* op = out + ((size_t)(bti * DM + co) << 12) + pin0;
    *(v4f*)(op)     = lo;
    *(v4f*)(op + 4) = hi4;
  }
}

// ---------------------------------------------------------------------------
extern "C" void kernel_launch(void* const* d_in, const int* in_sizes, int n_in,
                              void* d_out, int out_size, void* d_ws, size_t ws_size,
                              hipStream_t stream) {
  const float* x  = (const float*)d_in[0];
  const float* wq = (const float*)d_in[1];
  const float* bq = (const float*)d_in[2];
  const float* wk = (const float*)d_in[3];
  const float* bk = (const float*)d_in[4];
  const float* wv = (const float*)d_in[5];
  const float* bv = (const float*)d_in[6];
  const float* wo = (const float*)d_in[7];
  const float* bo = (const float*)d_in[8];
  float* out = (float*)d_out;

  // workspace: Q | K | V^T | S(=P) | XT | WQKV | WOT ; Y(NHWC) aliases dead Q
  char* ws = (char*)d_ws;
  const size_t QE = (size_t)BT * DM * PIX;            // 16,777,216 elems
  const size_t QB = QE * sizeof(__bf16);              // 33.5 MB
  const size_t SB = (size_t)BATCH * (512 * 512 + 2048 * 2048) * sizeof(__bf16);
  __bf16* qbuf  = (__bf16*)(ws);
  __bf16* kbuf  = (__bf16*)(ws + QB);
  __bf16* vtbuf = (__bf16*)(ws + 2 * QB);
  __bf16* sbuf  = (__bf16*)(ws + 3 * QB);             // 17.8 MB
  __bf16* xt    = (__bf16*)(ws + 3 * QB + SB);        // 16.8 MB
  __bf16* wqkv  = (__bf16*)(ws + 3 * QB + SB + (size_t)BT * PIX * CI * 2);
  __bf16* wot   = (__bf16*)((char*)wqkv + (size_t)3 * DM * CI * 2);
  __bf16* ybf   = qbuf;                               // Q dead after scores
  const size_t C1 = (size_t)BATCH * 512 * 8192;       // chunk1 base in Q/K/V^T
  const size_t S1 = (size_t)BATCH * 512 * 512;        // chunk1 base in S

  // prep
  prep_weights_kernel<<<(9 * DM * DM + 255) / 256, 256, 0, stream>>>(wq, wk, wv, wo, wqkv, wot);
  xt_kernel<<<4 * 2048, dim3(32, 8), 0, stream>>>(x, xt);
  // 1) QKV projection
  {
    const int waves = 3 * 16 * 2048;
    qkv_proj_kernel<<<(waves + 7) / 8, 256, 0, stream>>>(xt, wqkv, bq, bk, bv, qbuf, kbuf, vtbuf);
  }
  // 2) scores per chunk (S^T-tiled, 4-way)
  {
    const int waves = BATCH * 32 * 8;
    attn_scores_kernel<<<(waves + 7) / 8, 256, 0, stream>>>(qbuf, kbuf, sbuf, 512, 8192);
  }
  {
    const int waves = BATCH * 128 * 32;
    attn_scores_kernel<<<(waves + 7) / 8, 256, 0, stream>>>(qbuf + C1, kbuf + C1, sbuf + S1, 2048, 2048);
  }
  // 3) softmax (in place)
  softmax_kernel<<<BATCH * 512, 256, 0, stream>>>(sbuf, 512);
  softmax_kernel<<<BATCH * 2048, 256, 0, stream>>>(sbuf + S1, 2048);
  // 4) P.V -> NHWC bf16 (4-way)
  {
    const int waves = BATCH * 512 * 8;
    attn_pv_kernel<<<(waves + 7) / 8, 256, 0, stream>>>(sbuf, vtbuf, ybf, 512, 8192, 0);
  }
  {
    const int waves = BATCH * 128 * 32;
    attn_pv_kernel<<<(waves + 7) / 8, 256, 0, stream>>>(sbuf + S1, vtbuf + C1, ybf, 2048, 2048, 1);
  }
  // 5) 3x3 conv + bias + LeakyReLU (4-way)
  {
    const int waves = 4096 * 4;
    conv3x3_kernel<<<(waves + 7) / 8, 256, 0, stream>>>(ybf, wot, bo, out);
  }
}
